// DeformConvWarp_5695126635154
// MI455X (gfx1250) — matrix operand order repaired and checked by reading, not compiled, not run on hardware
//
#include <hip/hip_runtime.h>
#include <hip/hip_bf16.h>

// ---------------------------------------------------------------------------
// Deformable conv2d for MI455X (gfx1250):
//   phase A: bilinear im2col gather -> bf16 patches tile in LDS
//   phase B: v_wmma_f32_16x16x32_bf16 GEMM over C*K = 1152 contraction
//   phase C: LDS transpose -> coalesced vector stores
// Shapes: x[4,128,128,128] f32, offset[4,18,128,128] f32, weight[128,128,9] f32
// out[4,128,128,128] f32.  GEMM: M = B*Ho*Wo = 65536, N = O = 128, K = 1152.
// ---------------------------------------------------------------------------

typedef __attribute__((ext_vector_type(16))) __bf16 v16bf;
typedef __attribute__((ext_vector_type(8)))  __bf16 v8bf;
typedef __attribute__((ext_vector_type(8)))  float  v8f;
typedef __attribute__((ext_vector_type(4)))  float  v4f;

#define C_IN   128
#define O_OUT  128
#define HW     128
#define KTAPS  9
#define CK     (C_IN * KTAPS)     // 1152 contraction depth
#define KKB    (CK / 32)          // 36 wmma steps of depth 32
#define TILE_M 16                 // spatial positions per workgroup

union AFrag { v16bf v; v8bf h[2]; };
union DFrag { v8f   v; v4f  h[2]; };

__device__ __forceinline__ unsigned short f2bf_rne(float f) {
    unsigned int u = __float_as_uint(f);
    unsigned int r = u + 0x7FFFu + ((u >> 16) & 1u);   // round-to-nearest-even
    return (unsigned short)(r >> 16);
}

// ---------------------------------------------------------------------------
// Kernel 0: pack weight[o][c][k] (f32) into WMMA B-fragment layout (bf16).
// Fragment (oB in 0..7, kkB in 0..35): lane L in 0..31 holds column N = oB*16 + L%16;
// its 16 contiguous bf16 hold K = kkB*32 + {8h..8h+7, 16+8h..23+8h}, h = L/16.  1 KB per fragment.
// ---------------------------------------------------------------------------
__global__ void dcw_pack_weights(const float* __restrict__ w,
                                 unsigned short* __restrict__ wp) {
    int idx = blockIdx.x * blockDim.x + threadIdx.x;   // 128*1152 elements
    if (idx >= O_OUT * CK) return;
    int o = idx / CK;
    int r = idx - o * CK;          // r = c*9 + k  (contraction index)
    int oB   = o >> 4;
    int n    = o & 15;
    int kkB  = r >> 5;
    int lane = r & 31;
    wp[(((oB * KKB) + kkB) * 32 + (((lane >> 3) & 1) << 4) + n) * 16 + (lane & 7) + ((lane >> 4) << 3)] = f2bf_rne(w[idx]);
}

// ---------------------------------------------------------------------------
// Kernel 1: one workgroup per (b, ho, wo-block-of-16).  256 threads = 8 waves.
// ---------------------------------------------------------------------------
__global__ void __launch_bounds__(256, 2)
dcw_main(const float* __restrict__ x,
         const float* __restrict__ off,
         const unsigned short* __restrict__ wp,
         float* __restrict__ out) {
    __shared__ unsigned short sPatch[TILE_M * CK];    // 36 KB bf16 patches
    __shared__ float sStage[8 * 16 * 16];             // 8 KB f32 D staging

    const int bx = blockIdx.x;                        // 4096 blocks
    const int woB = (bx & 7) << 4;                    // wo base (0..112, x16)
    const int ho  = (bx >> 3) & (HW - 1);
    const int b   = bx >> 10;

    const int tid  = threadIdx.x;
    const int wave = tid >> 5;
    const int lane = tid & 31;

    // Warm L2 with this wave's weight panel while the gather runs.
    __builtin_prefetch(wp + (size_t)wave * KKB * 32 * 16, 0, 1);

    // ---------------- Phase A: bilinear gather into LDS (bf16) -------------
    {
        const int m  = tid >> 4;          // spatial position in tile, 0..15
        const int cg = tid & 15;          // channel group, 0..15
        const int wo = woB + m;
        const float* offb = off + (((size_t)b * 2 * KTAPS) * HW + ho) * HW + wo;

        #pragma unroll
        for (int k = 0; k < KTAPS; ++k) {
            const int ky = (k / 3) - 1;   // pad=1, dil=1, stride=1
            const int kx = (k % 3) - 1;
            const float oy = offb[(size_t)(2 * k)     * HW * HW];
            const float ox = offb[(size_t)(2 * k + 1) * HW * HW];
            const float py = (float)(ho + ky) + oy;
            const float px = (float)(wo + kx) + ox;
            const float fy = floorf(py), fx = floorf(px);
            const int   y0 = (int)fy,   x0 = (int)fx;
            const float wy = py - fy,   wx = px - fx;

            const float vy0 = (y0 >= 0 && y0 < HW)     ? 1.f : 0.f;
            const float vy1 = (y0 + 1 >= 0 && y0 + 1 < HW) ? 1.f : 0.f;
            const float vx0 = (x0 >= 0 && x0 < HW)     ? 1.f : 0.f;
            const float vx1 = (x0 + 1 >= 0 && x0 + 1 < HW) ? 1.f : 0.f;
            const float w00 = (1.f - wy) * (1.f - wx) * vy0 * vx0;
            const float w01 = (1.f - wy) * wx         * vy0 * vx1;
            const float w10 = wy * (1.f - wx)         * vy1 * vx0;
            const float w11 = wy * wx                 * vy1 * vx1;

            const int cy0 = min(max(y0, 0), HW - 1);
            const int cy1 = min(max(y0 + 1, 0), HW - 1);
            const int cx0 = min(max(x0, 0), HW - 1);
            const int cx1 = min(max(x0 + 1, 0), HW - 1);
            const int i00 = cy0 * HW + cx0;
            const int i01 = cy0 * HW + cx1;
            const int i10 = cy1 * HW + cx0;
            const int i11 = cy1 * HW + cx1;

            #pragma unroll
            for (int ci = 0; ci < 8; ++ci) {
                const int c = cg + (ci << 4);
                const float* xb = x + (((size_t)b * C_IN + c) << 14);
                float v = w00 * xb[i00];
                v = fmaf(w01, xb[i01], v);
                v = fmaf(w10, xb[i10], v);
                v = fmaf(w11, xb[i11], v);
                sPatch[m * CK + c * KTAPS + k] = f2bf_rne(v);
            }
        }
    }
    __syncthreads();

    // ---------------- Phase B: WMMA GEMM over 1152-deep contraction --------
    // Wave `wave` computes 16 spatial rows x output channels [16*wave, +16).
    const int mrow  = lane & 15;
    const int khalf = (lane >> 4) << 3;          // 0 (lanes 0-15) or 8
    const unsigned short* bpanel = wp + (size_t)wave * KKB * 32 * 16;

    v8f acc = {};
    #pragma unroll 4
    for (int kk = 0; kk < KKB; ++kk) {
        // A fragment (16x32 bf16): lane L<16 -> row M=L, K in {0..7,16..23};
        // lane L>=16 -> row M=L-16, K in {8..15,24..31}.
        const unsigned short* ap = &sPatch[mrow * CK + (kk << 5) + khalf];
        AFrag a;
        a.h[0] = *(const v8bf*)(ap);
        a.h[1] = *(const v8bf*)(ap + 16);

        // B fragment (32x16 bf16): lane L -> column N=L%16, 16 K values in the same order as A.
        const unsigned short* bp = bpanel + ((kk << 5) + lane) * 16;
        AFrag bm;
        bm.h[0] = *(const v8bf*)(bp);
        bm.h[1] = *(const v8bf*)(bp + 8);

        acc = __builtin_amdgcn_wmma_f32_16x16x32_bf16(
            /*neg_a=*/false, a.v, /*neg_b=*/false, bm.v,
            /*c_mod=*/(short)0, acc, /*reuse_a=*/false, /*reuse_b=*/false);
    }

    // ---------------- Phase C: transpose via LDS, coalesced stores ---------
    // D layout: lanes 0-15 hold N=lane, VGPR v -> M=v; lanes 16-31: M=v+8.
    float* st = &sStage[wave * 256];
    {
        DFrag d; d.v = acc;
        const int mb = (lane >> 4) << 3;
        v4f* dst = (v4f*)&st[(lane & 15) * 16 + mb];   // stage[o=N][m]
        dst[0] = d.h[0];
        dst[1] = d.h[1];
    }
    __syncthreads();
    {
        const int o = (wave << 4) + (lane >> 1);
        const int m = (lane & 1) << 3;
        const v4f* src = (const v4f*)&st[lane * 8];
        float* op = out + (((size_t)b * O_OUT + o) << 14) + ho * HW + woB + m;
        ((v4f*)op)[0] = src[0];
        ((v4f*)op)[1] = src[1];
    }
}

// ---------------------------------------------------------------------------
extern "C" void kernel_launch(void* const* d_in, const int* in_sizes, int n_in,
                              void* d_out, int out_size, void* d_ws, size_t ws_size,
                              hipStream_t stream) {
    const float* x   = (const float*)d_in[0];   // [4,128,128,128]
    const float* off = (const float*)d_in[1];   // [4,18,128,128]
    const float* w   = (const float*)d_in[2];   // [128,128,3,3]
    float* out = (float*)d_out;
    unsigned short* wp = (unsigned short*)d_ws; // 128*1152 bf16 = 288 KB

    // Kernel 0: pack weights into WMMA B-fragment layout.
    {
        int n = O_OUT * CK;
        dcw_pack_weights<<<(n + 255) / 256, 256, 0, stream>>>(w, wp);
    }
    // Kernel 1: 4096 tiles of (b, ho, 16-wide wo block).
    {
        int blocks = 4 * HW * (HW / TILE_M);    // 4096
        dcw_main<<<blocks, 256, 0, stream>>>(x, off, wp, out);
    }
}
